// GnBlock_15212774163064
// MI455X (gfx1250) — matrix operand order repaired and checked by reading, not compiled, not run on hardware
//
#include <hip/hip_runtime.h>

// GN message-passing block for MI455X (gfx1250, wave32).
// All GEMMs via v_wmma_f32_16x16x32_bf16 (f32 data, bf16 operands, f32 accum).
// Workspace layout (d_ws, needs ~52 MB):
//   [0, N*D)            f32 agg (segment-sum of edge messages)
//   [N*D, N*D+512)      f32 stats: colsum[128], colsumsq[128], mean[128], invrms[1]
//   then bf16 transposed weights: W1E[128][384], W2E[128][128], W1N[128][256], W2N[128][128]

#define D 128
#define KE 384   // edge MLP layer-1 K (3*D)
#define KN 256   // node MLP layer-1 K (2*D)

typedef float v8f   __attribute__((ext_vector_type(8)));
typedef __bf16 v16bf __attribute__((ext_vector_type(16)));
typedef unsigned int u32x4 __attribute__((ext_vector_type(4)));
typedef float f32x4 __attribute__((ext_vector_type(4)));

union ABfrag { v16bf v; unsigned int u[8]; u32x4 q[2]; };

// Fast truncating f32x2 -> packed bf16x2: one v_perm_b32 (high halves of both floats).
__device__ __forceinline__ unsigned int pack2_bf16(float a, float b) {
  return __builtin_amdgcn_perm(__builtin_bit_cast(unsigned int, b),
                               __builtin_bit_cast(unsigned int, a),
                               0x07060302u);
}
// Truncating scalar f32 -> bf16 (used for the LDS h1 staging stores).
__device__ __forceinline__ unsigned short f32_to_bf16_fast(float a) {
  return (unsigned short)(__builtin_bit_cast(unsigned int, a) >> 16);
}
// RNE f32 -> bf16 (one-time weight prep only).
__device__ __forceinline__ unsigned short f32_to_bf16_rne(float a) {
  unsigned int ua = __builtin_bit_cast(unsigned int, a);
  ua = (ua + 0x7FFFu + ((ua >> 16) & 1u)) >> 16;
  return (unsigned short)ua;
}

// ---------------------------------------------------------------- zero scratch
__global__ void gn_zero_kernel(float* __restrict__ p, long n) {
  long stride = (long)gridDim.x * blockDim.x;
  for (long i = (long)blockIdx.x * blockDim.x + threadIdx.x; i < n; i += stride)
    p[i] = 0.0f;
}

// ------------------------------------------------- f32 weights -> bf16 W^T[n][k]
__global__ void gn_wprep_kernel(const float* __restrict__ eW1, const float* __restrict__ eW2,
                                const float* __restrict__ nW1, const float* __restrict__ nW2,
                                unsigned short* __restrict__ W1E, unsigned short* __restrict__ W2E,
                                unsigned short* __restrict__ W1N, unsigned short* __restrict__ W2N) {
  const int total = 128 * KE + 128 * D + 128 * KN + 128 * D;  // 114688
  int stride = gridDim.x * blockDim.x;
  for (int t = blockIdx.x * blockDim.x + threadIdx.x; t < total; t += stride) {
    int u = t;
    if (u < 128 * KE) { int n = u / KE, k = u % KE; W1E[u] = f32_to_bf16_rne(eW1[k * D + n]); continue; }
    u -= 128 * KE;
    if (u < 128 * D)  { int n = u / D,  k = u % D;  W2E[u] = f32_to_bf16_rne(eW2[k * D + n]); continue; }
    u -= 128 * D;
    if (u < 128 * KN) { int n = u / KN, k = u % KN; W1N[u] = f32_to_bf16_rne(nW1[k * D + n]); continue; }
    u -= 128 * KN;
    { int n = u / D, k = u % D; W2N[u] = f32_to_bf16_rne(nW2[k * D + n]); }
  }
}

// -------------------------------------------------------------------- edge MLP
// One wave == one 16-edge tile. Layer1: [16x384]@[384x128], Layer2: [16x128]@[128x128].
__global__ __launch_bounds__(256) void gn_edge_kernel(
    const float* __restrict__ x, const float* __restrict__ edge_attr,
    const int* __restrict__ eidx,
    const float* __restrict__ eb1, const float* __restrict__ eb2,
    const float* __restrict__ eg,  const float* __restrict__ ebt,
    const unsigned short* __restrict__ W1t, const unsigned short* __restrict__ W2t,
    float* __restrict__ agg, float* __restrict__ edge_out, int E) {
  __shared__ unsigned short hs[8][16 * D];        // per-wave h1 staging (bf16, row-major)

  const int wave = threadIdx.x >> 5;
  const int lane = threadIdx.x & 31;
  const int l15  = lane & 15;
  const int half = lane >> 4;
  const int tile = blockIdx.x * 8 + wave;
  if (tile * 16 >= E) return;

  const long eA  = (long)tile * 16 + l15;         // edge whose row this lane loads
  const int  src = eidx[eA];
  const int  dst = eidx[(long)E + eA];
  const float* xsrc  = x + (long)src * D;
  const float* xdst  = x + (long)dst * D;
  const float* earow = edge_attr + eA * D;
  __builtin_prefetch(earow, 0, 1);                // global_prefetch into L2

  // ---- layer 1: K = 384 in 12 chunks of 32 --------------------------------
  v8f acc[8] = {};
  for (int kc = 0; kc < 12; ++kc) {
    const int col0   = kc * 32 + half * 8;        // chunk stays inside one concat region
    const int region = col0 >> 7;
    const float* rp  = (region == 0) ? xsrc : (region == 1) ? xdst : earow;
    const int c      = col0 & 127;
    f32x4 f0 = *(const f32x4*)(rp + c);
    f32x4 f1 = *(const f32x4*)(rp + c + 4);
    f32x4 f2 = *(const f32x4*)(rp + c + 16);
    f32x4 f3 = *(const f32x4*)(rp + c + 20);
    ABfrag A;
    A.u[0] = pack2_bf16(f0.x, f0.y); A.u[1] = pack2_bf16(f0.z, f0.w);
    A.u[2] = pack2_bf16(f1.x, f1.y); A.u[3] = pack2_bf16(f1.z, f1.w);
    A.u[4] = pack2_bf16(f2.x, f2.y); A.u[5] = pack2_bf16(f2.z, f2.w);
    A.u[6] = pack2_bf16(f3.x, f3.y); A.u[7] = pack2_bf16(f3.z, f3.w);
    const int kb2 = kc * 32 + half * 8;           // B-fragment: K kb2..kb2+7 and kb2+16..kb2+23 per lane
#pragma unroll
    for (int nt = 0; nt < 8; ++nt) {
      const int n = nt * 16 + l15;
      const u32x4* bp = (const u32x4*)(W1t + n * KE + kb2);
      ABfrag B; B.q[0] = bp[0]; B.q[1] = bp[2];
      acc[nt] = __builtin_amdgcn_wmma_f32_16x16x32_bf16(
          false, A.v, false, B.v, (short)0, acc[nt], false, false);
    }
  }

  // ---- bias + ReLU -> bf16 h1 in LDS (row-major [m][k]) -------------------
  unsigned short* h = hs[wave];
#pragma unroll
  for (int nt = 0; nt < 8; ++nt) {
    const float b1 = eb1[nt * 16 + l15];
#pragma unroll
    for (int r = 0; r < 8; ++r) {
      float v = acc[nt][r] + b1;
      v = v > 0.0f ? v : 0.0f;
      h[(r + half * 8) * D + nt * 16 + l15] = f32_to_bf16_fast(v);
    }
  }
  asm volatile("s_wait_dscnt 0" ::: "memory");    // LDS store -> same-wave load hazard

  // ---- layer 2: K = 128 in 4 chunks of 32 ---------------------------------
  v8f acc2[8] = {};
  for (int kc = 0; kc < 4; ++kc) {
    const int kb = kc * 32 + half * 8;
    ABfrag A;
    A.q[0] = *(const u32x4*)(h + l15 * D + kb);
    A.q[1] = *(const u32x4*)(h + l15 * D + kb + 16);
    const int kb2 = kc * 32 + half * 8;
#pragma unroll
    for (int nt = 0; nt < 8; ++nt) {
      const int n = nt * 16 + l15;
      const u32x4* bp = (const u32x4*)(W2t + n * D + kb2);
      ABfrag B; B.q[0] = bp[0]; B.q[1] = bp[2];
      acc2[nt] = __builtin_amdgcn_wmma_f32_16x16x32_bf16(
          false, A.v, false, B.v, (short)0, acc2[nt], false, false);
    }
  }

  // ---- bias + LayerNorm + residual store + scatter-add --------------------
  float g[8], bt[8], b2[8];
#pragma unroll
  for (int nt = 0; nt < 8; ++nt) {
    const int n = nt * 16 + l15;
    g[nt] = eg[n]; bt[nt] = ebt[n]; b2[nt] = eb2[n];
  }
#pragma unroll
  for (int r = 0; r < 8; ++r) {
    const int m = r + half * 8;                   // row (edge) for this half-lane
    float v[8], s = 0.0f, sq = 0.0f;
#pragma unroll
    for (int nt = 0; nt < 8; ++nt) {
      v[nt] = acc2[nt][r] + b2[nt];
      s += v[nt]; sq += v[nt] * v[nt];
    }
    for (int o = 1; o < 16; o <<= 1) {            // reduce across the 16-lane half
      s  += __shfl_xor(s,  o, 32);
      sq += __shfl_xor(sq, o, 32);
    }
    const float mu   = s * (1.0f / 128.0f);
    const float var  = sq * (1.0f / 128.0f) - mu * mu;
    const float rstd = rsqrtf(var + 1e-5f);
    const int   dm   = __shfl(dst, m, 32);        // dst node of edge (tile*16+m)
    const long  erow = ((long)tile * 16 + m) * D;
#pragma unroll
    for (int nt = 0; nt < 8; ++nt) {
      const float o = (v[nt] - mu) * rstd * g[nt] + bt[nt];
      const int   n = nt * 16 + l15;
      edge_out[erow + n] = edge_attr[erow + n] + o;   // edge residual
      atomicAdd(agg + (long)dm * D + n, o);           // segment-sum
    }
  }
}

// -------------------------------------------------------------------- node MLP
__global__ __launch_bounds__(256) void gn_node_kernel(
    const float* __restrict__ x, const float* __restrict__ agg,
    const float* __restrict__ nb1, const float* __restrict__ nb2,
    const float* __restrict__ ng,  const float* __restrict__ nbt,
    const unsigned short* __restrict__ W1t, const unsigned short* __restrict__ W2t,
    float* __restrict__ x_pre, int N) {
  __shared__ unsigned short hs[8][16 * D];

  const int wave = threadIdx.x >> 5;
  const int lane = threadIdx.x & 31;
  const int l15  = lane & 15;
  const int half = lane >> 4;
  const int tile = blockIdx.x * 8 + wave;
  if (tile * 16 >= N) return;

  const long node  = (long)tile * 16 + l15;
  const float* xr  = x + node * D;
  const float* ar  = agg + node * D;

  v8f acc[8] = {};
  for (int kc = 0; kc < 8; ++kc) {                // K = 256 (x || agg)
    const int col0   = kc * 32 + half * 8;
    const float* rp  = (col0 < 128) ? xr : ar;
    const int c      = col0 & 127;
    f32x4 f0 = *(const f32x4*)(rp + c);
    f32x4 f1 = *(const f32x4*)(rp + c + 4);
    f32x4 f2 = *(const f32x4*)(rp + c + 16);
    f32x4 f3 = *(const f32x4*)(rp + c + 20);
    ABfrag A;
    A.u[0] = pack2_bf16(f0.x, f0.y); A.u[1] = pack2_bf16(f0.z, f0.w);
    A.u[2] = pack2_bf16(f1.x, f1.y); A.u[3] = pack2_bf16(f1.z, f1.w);
    A.u[4] = pack2_bf16(f2.x, f2.y); A.u[5] = pack2_bf16(f2.z, f2.w);
    A.u[6] = pack2_bf16(f3.x, f3.y); A.u[7] = pack2_bf16(f3.z, f3.w);
    const int kb2 = kc * 32 + half * 8;
#pragma unroll
    for (int nt = 0; nt < 8; ++nt) {
      const int n = nt * 16 + l15;
      const u32x4* bp = (const u32x4*)(W1t + n * KN + kb2);
      ABfrag B; B.q[0] = bp[0]; B.q[1] = bp[2];
      acc[nt] = __builtin_amdgcn_wmma_f32_16x16x32_bf16(
          false, A.v, false, B.v, (short)0, acc[nt], false, false);
    }
  }

  unsigned short* h = hs[wave];
#pragma unroll
  for (int nt = 0; nt < 8; ++nt) {
    const float b1 = nb1[nt * 16 + l15];
#pragma unroll
    for (int r = 0; r < 8; ++r) {
      float v = acc[nt][r] + b1;
      v = v > 0.0f ? v : 0.0f;
      h[(r + half * 8) * D + nt * 16 + l15] = f32_to_bf16_fast(v);
    }
  }
  asm volatile("s_wait_dscnt 0" ::: "memory");

  v8f acc2[8] = {};
  for (int kc = 0; kc < 4; ++kc) {
    const int kb = kc * 32 + half * 8;
    ABfrag A;
    A.q[0] = *(const u32x4*)(h + l15 * D + kb);
    A.q[1] = *(const u32x4*)(h + l15 * D + kb + 16);
    const int kb2 = kc * 32 + half * 8;
#pragma unroll
    for (int nt = 0; nt < 8; ++nt) {
      const int n = nt * 16 + l15;
      const u32x4* bp = (const u32x4*)(W2t + n * D + kb2);
      ABfrag B; B.q[0] = bp[0]; B.q[1] = bp[2];
      acc2[nt] = __builtin_amdgcn_wmma_f32_16x16x32_bf16(
          false, A.v, false, B.v, (short)0, acc2[nt], false, false);
    }
  }

  float g[8], bt[8], b2[8];
#pragma unroll
  for (int nt = 0; nt < 8; ++nt) {
    const int n = nt * 16 + l15;
    g[nt] = ng[n]; bt[nt] = nbt[n]; b2[nt] = nb2[n];
  }
#pragma unroll
  for (int r = 0; r < 8; ++r) {
    const int m = r + half * 8;
    float v[8], s = 0.0f, sq = 0.0f;
#pragma unroll
    for (int nt = 0; nt < 8; ++nt) {
      v[nt] = acc2[nt][r] + b2[nt];
      s += v[nt]; sq += v[nt] * v[nt];
    }
    for (int o = 1; o < 16; o <<= 1) {
      s  += __shfl_xor(s,  o, 32);
      sq += __shfl_xor(sq, o, 32);
    }
    const float mu   = s * (1.0f / 128.0f);
    const float var  = sq * (1.0f / 128.0f) - mu * mu;
    const float rstd = rsqrtf(var + 1e-5f);
    const long  nrow = ((long)tile * 16 + m) * D;
#pragma unroll
    for (int nt = 0; nt < 8; ++nt) {
      const float o = (v[nt] - mu) * rstd * g[nt] + bt[nt];
      const int   n = nt * 16 + l15;
      x_pre[nrow + n] = x[nrow + n] + o;          // node residual (pre-PairNorm)
    }
  }
}

// --------------------------------------------------- PairNorm: column statistics
__global__ void gn_colstat_kernel(const float* __restrict__ xp,
                                  float* __restrict__ colsum, float* __restrict__ colsumsq,
                                  int N) {
  const int c = threadIdx.x & 127;
  const int r0 = blockIdx.x * 2 + (threadIdx.x >> 7);
  const int rs = gridDim.x * 2;
  float s = 0.0f, sq = 0.0f;
  for (int r = r0; r < N; r += rs) {
    float v = xp[(long)r * D + c];
    s += v; sq += v * v;
  }
  atomicAdd(&colsum[c], s);
  atomicAdd(&colsumsq[c], sq);
}

__global__ void gn_pairnorm_finalize(const float* __restrict__ colsum,
                                     const float* __restrict__ colsumsq,
                                     float* __restrict__ mean, float* __restrict__ invrms,
                                     int N) {
  __shared__ float red[128];
  const int c = threadIdx.x;                      // 128 threads
  const float fn = (float)N;
  const float mu = colsum[c] / fn;
  mean[c] = mu;
  red[c] = colsumsq[c] - fn * mu * mu;            // sum (x - mu_c)^2 for this column
  __syncthreads();
  for (int o = 64; o > 0; o >>= 1) {
    if (c < o) red[c] += red[c + o];
    __syncthreads();
  }
  if (c == 0) {
    const float rms = sqrtf(red[0] / fn) + 1e-8f;
    invrms[0] = 1.0f / rms;
  }
}

__global__ void gn_pairnorm_apply(float* __restrict__ xo, const float* __restrict__ mean,
                                  const float* __restrict__ invrms, long total4) {
  const float ir = invrms[0];
  const long stride = (long)gridDim.x * blockDim.x;
  for (long i = (long)blockIdx.x * blockDim.x + threadIdx.x; i < total4; i += stride) {
    f32x4 v = ((f32x4*)xo)[i];
    const f32x4 mu = *(const f32x4*)(mean + ((i & 31) << 2));
    v = (v - mu) * ir;
    ((f32x4*)xo)[i] = v;
  }
}

// ------------------------------------------------------------------------------
extern "C" void kernel_launch(void* const* d_in, const int* in_sizes, int n_in,
                              void* d_out, int out_size, void* d_ws, size_t ws_size,
                              hipStream_t stream) {
  const float* x         = (const float*)d_in[0];
  const float* edge_attr = (const float*)d_in[1];
  const int*   eidx      = (const int*)d_in[2];
  const float* eW1 = (const float*)d_in[3];
  const float* eb1 = (const float*)d_in[4];
  const float* eW2 = (const float*)d_in[5];
  const float* eb2 = (const float*)d_in[6];
  const float* eg  = (const float*)d_in[7];
  const float* ebt = (const float*)d_in[8];
  const float* nW1 = (const float*)d_in[9];
  const float* nb1 = (const float*)d_in[10];
  const float* nW2 = (const float*)d_in[11];
  const float* nb2 = (const float*)d_in[12];
  const float* ng  = (const float*)d_in[13];
  const float* nbt = (const float*)d_in[14];

  const int N = in_sizes[0] / D;   // 100000
  const int E = in_sizes[1] / D;   // 600000

  float* out_x = (float*)d_out;                    // [N, D]
  float* out_e = (float*)d_out + (long)N * D;      // [E, D]

  // workspace carve-out
  float* agg      = (float*)d_ws;                  // [N, D]
  float* stats    = agg + (long)N * D;
  float* colsum   = stats;
  float* colsumsq = stats + 128;
  float* mean     = stats + 256;
  float* invrms   = stats + 384;
  unsigned short* W1E = (unsigned short*)(stats + 512);
  unsigned short* W2E = W1E + 128 * KE;
  unsigned short* W1N = W2E + 128 * D;
  unsigned short* W2N = W1N + 128 * KN;

  // 1. zero agg + colsum/colsumsq
  gn_zero_kernel<<<2048, 256, 0, stream>>>(agg, (long)N * D + 256);
  // 2. f32 weights -> bf16 transposed layouts
  gn_wprep_kernel<<<448, 256, 0, stream>>>(eW1, eW2, nW1, nW2, W1E, W2E, W1N, W2N);
  // 3. edge MLP + scatter-add (one wave per 16 edges, 8 waves per block)
  {
    const int tiles = (E + 15) / 16;
    gn_edge_kernel<<<(tiles + 7) / 8, 256, 0, stream>>>(
        x, edge_attr, eidx, eb1, eb2, eg, ebt, W1E, W2E, agg, out_e, E);
  }
  // 4. node MLP -> pre-PairNorm x into d_out
  {
    const int tiles = (N + 15) / 16;
    gn_node_kernel<<<(tiles + 7) / 8, 256, 0, stream>>>(
        x, agg, nb1, nb2, ng, nbt, W1N, W2N, out_x, N);
  }
  // 5-7. PairNorm
  gn_colstat_kernel<<<1024, 256, 0, stream>>>(out_x, colsum, colsumsq, N);
  gn_pairnorm_finalize<<<1, 128, 0, stream>>>(colsum, colsumsq, mean, invrms, N);
  gn_pairnorm_apply<<<2048, 256, 0, stream>>>(out_x, mean, invrms, (long)N * (D / 4));
}
